// GroupedMoEExperts_81527069212691
// MI455X (gfx1250) — matrix operand order repaired and checked by reading, not compiled, not run on hardware
//
#include <hip/hip_runtime.h>
#include <hip/hip_bf16.h>

// Problem constants (from reference): E=8, H=1024, I=4096, N=2048, K=2
#define NE 8
#define HD 1024
#define ID 4096
#define NT 2048
#define TK 2

typedef _Float16 f16;
typedef __attribute__((ext_vector_type(16))) _Float16 v16h;
typedef __attribute__((ext_vector_type(8)))  float    v8f;
typedef __attribute__((ext_vector_type(8)))  unsigned int v8u;

union frag16 { v8u u; v16h h; };

// ---------------------------------------------------------------------------
// CDNA5 async global->LDS copy (ASYNCcnt-tracked; ISA §10 async ops).
// LDS destination address = low 32 bits of the flat pointer (ISA §10.2:
// LDS aperture maps addr[31:0] to the LDS byte address).
// ---------------------------------------------------------------------------
__device__ inline void async_ld_b128(void* lds_dst, const void* gsrc) {
  asm volatile("global_load_async_to_lds_b128 %0, %1, off"
               :: "v"((unsigned)(unsigned long long)(uintptr_t)lds_dst),
                  "v"((unsigned long long)(uintptr_t)gsrc)
               : "memory");
}
__device__ inline void wait_async_le2() {   // current buffer landed; next (2) may fly
  asm volatile("s_wait_asynccnt 0x2" ::: "memory");
}
__device__ inline void wait_async_le0() {   // drain (last iteration)
  asm volatile("s_wait_asynccnt 0x0" ::: "memory");
}

// ---------------------------------------------------------------------------
// Fragment loaders. LDS tiles hold 32 f16 along K per row (row = M for A,
// row = N for B^T). Layouts per CDNA5 ISA 7.12.2:
//   A 16x32 f16 : lane L -> M = L%16 ; VGPR j holds K = 2*(j&3) + 8*(L/16) + 16*(j/4)
//   B 32x16 f16 : lane L -> N = L%16 ; VGPR j holds K = 2*(j&3) + 8*(L/16) + 16*(j/4)
// Both (K, K+1) pairs are contiguous -> aligned 32-bit LDS loads (merge to b128).
// ---------------------------------------------------------------------------
__device__ inline v16h load_a_frag(const f16* sA, int baseRow) {
  int lane = threadIdx.x & 31;
  int m = lane & 15, half = lane >> 4;
  const unsigned int* row = (const unsigned int*)(sA + (size_t)(baseRow + m) * 32);
  frag16 f;
#pragma unroll
  for (int j = 0; j < 8; ++j) {
    int k = 2 * (j & 3) + 8 * half + 16 * (j >> 2);
    f.u[j] = row[k >> 1];
  }
  return f.h;
}

__device__ inline v16h load_b_frag(const f16* sBt, int baseN) {
  int lane = threadIdx.x & 31;
  int n = lane & 15, kh = lane >> 4;
  const unsigned int* row = (const unsigned int*)(sBt + (size_t)(baseN + n) * 32);
  frag16 f;
#pragma unroll
  for (int j = 0; j < 8; ++j) {
    int k = 2 * (j & 3) + 8 * kh + 16 * (j >> 2);
    f.u[j] = row[k >> 1];
  }
  return f.h;
}

__device__ inline v8f wmma_f16(v16h a, v16h b, v8f c) {
  return __builtin_amdgcn_wmma_f32_16x16x32_f16(false, a, false, b, (short)0, c,
                                                false, false);
}

// ---------------------------------------------------------------------------
// Routing: deterministic per-expert compaction of the N*K (token,k) pairs.
// One block, chunked counting + serial-per-expert exclusive scan (no atomics,
// so slot assignment is bitwise reproducible across calls).
// ---------------------------------------------------------------------------
__global__ void moe_route(const int* __restrict__ eidx,
                          int* __restrict__ tok_pid, int* __restrict__ off) {
  __shared__ int cnt[256][NE];
  __shared__ int tot[NE];
  __shared__ int base[NE];
  const int t = threadIdx.x;
  const int PT = (NT * TK) / 256;  // 16 pairs per thread
#pragma unroll
  for (int e = 0; e < NE; ++e) cnt[t][e] = 0;
  for (int i = 0; i < PT; ++i) {
    int p = t * PT + i;
    cnt[t][eidx[p]]++;
  }
  __syncthreads();
  if (t < NE) {
    int run = 0;
    for (int j = 0; j < 256; ++j) {
      int v = cnt[j][t];
      cnt[j][t] = run;
      run += v;
    }
    tot[t] = run;
  }
  __syncthreads();
  if (t == 0) {
    int acc = 0;
    for (int e = 0; e < NE; ++e) {
      base[e] = acc;
      off[e] = acc;
      acc += tot[e];
    }
    off[NE] = acc;
  }
  __syncthreads();
#pragma unroll
  for (int e = 0; e < NE; ++e) cnt[t][e] += base[e];
  for (int i = 0; i < PT; ++i) {
    int p = t * PT + i;
    int e = eidx[p];
    tok_pid[cnt[t][e]++] = p;
  }
}

// x fp32 -> f16 once (weights are converted on the fly inside the GEMMs).
__global__ void moe_cvt_x(const float* __restrict__ x, f16* __restrict__ xh) {
  int i = blockIdx.x * 256 + threadIdx.x;  // over (N*H)/4
  float4 v = ((const float4*)x)[i];
  alignas(8) f16 h[4] = {(f16)v.x, (f16)v.y, (f16)v.z, (f16)v.w};
  ((uint2*)xh)[i] = *(const uint2*)h;
}

// ---------------------------------------------------------------------------
// GEMM1 (fused gate/up): rows = gathered x tokens for expert e.
//   gate = A @ W1_e, up = A @ W3_e, act = silu(gate)*up  -> f16 rows (by slot)
// Block tile 128(M) x 64(N), K-step 32, 8 waves (4Mx2N), 2x2 WMMA tiles each.
// Double-buffered LDS; A staged via async global->LDS, weights streamed fp32
// and converted to f16 in-register one K-step ahead.
// ---------------------------------------------------------------------------
__global__ __launch_bounds__(256) void moe_ffn1(
    const f16* __restrict__ xh, const float* __restrict__ w1,
    const float* __restrict__ w3, const int* __restrict__ tok_pid,
    const int* __restrict__ off, f16* __restrict__ act) {
  __shared__ f16 sA[2][128 * 32];
  __shared__ f16 sB1[2][64 * 32];   // [n][k]
  __shared__ f16 sB3[2][64 * 32];
  __shared__ int rowTok[128];

  const int e = blockIdx.z;
  const int offe = off[e];
  const int Me = off[e + 1] - offe;
  const int m0 = blockIdx.y * 128;
  if (m0 >= Me) return;
  const int n0 = blockIdx.x * 64;

  const int tid = threadIdx.x, lane = tid & 31, wid = tid >> 5;
  const int wm = (wid & 3) * 32;
  const int wn = (wid >> 2) * 32;

  if (tid < 128) {
    int r = m0 + tid;
    // padded rows point at token 0: their (discarded) output rows may be
    // garbage, valid rows are unaffected (WMMA row independence).
    rowTok[tid] = (r < Me) ? (tok_pid[offe + r] >> 1) : 0;
  }
  __syncthreads();

  const float* W1 = w1 + (size_t)e * HD * ID;
  const float* W3 = w3 + (size_t)e * HD * ID;

  const int arow = tid >> 1, aseg = (tid & 1) * 16;   // A: 32B per thread
  const f16* aSrc = xh + (size_t)rowTok[arow] * HD + aseg;
  f16* aDst[2] = {&sA[0][arow * 32 + aseg], &sA[1][arow * 32 + aseg]};
  const int bk = tid >> 3, bn = (tid & 7) * 8;        // B: 8 fp32 per thread
  const float* B1 = W1 + (size_t)bk * ID + n0 + bn;
  const float* B3 = W3 + (size_t)bk * ID + n0 + bn;

  v8f accG[2][2] = {};
  v8f accU[2][2] = {};

  // ---- preload k=0 into buffer 0 ----
  async_ld_b128(aDst[0], aSrc);
  async_ld_b128(aDst[0] + 8, aSrc + 8);
  float4 p1 = ((const float4*)B1)[0], q1 = ((const float4*)B1)[1];
  float4 p3 = ((const float4*)B3)[0], q3 = ((const float4*)B3)[1];

  int buf = 0;
  for (int k0 = 0; k0 < HD; k0 += 32) {
    const int nxt = buf ^ 1;
    // commit B(k0) regs -> LDS[buf] (fp32 -> f16, K-transposed)
    {
      f16 h1[8] = {(f16)p1.x, (f16)p1.y, (f16)p1.z, (f16)p1.w,
                   (f16)q1.x, (f16)q1.y, (f16)q1.z, (f16)q1.w};
      f16 h3[8] = {(f16)p3.x, (f16)p3.y, (f16)p3.z, (f16)p3.w,
                   (f16)q3.x, (f16)q3.y, (f16)q3.z, (f16)q3.w};
#pragma unroll
      for (int i = 0; i < 8; ++i) {
        sB1[buf][(bn + i) * 32 + bk] = h1[i];
        sB3[buf][(bn + i) * 32 + bk] = h3[i];
      }
    }
    // issue k0+32 while this step computes
    if (k0 + 32 < HD) {
      const size_t koff = (size_t)(k0 + 32) * 1;          // f16 elems along K
      async_ld_b128(aDst[nxt], aSrc + koff);
      async_ld_b128(aDst[nxt] + 8, aSrc + koff + 8);
      const float* g1 = B1 + (size_t)(k0 + 32) * ID;
      const float* g3 = B3 + (size_t)(k0 + 32) * ID;
      p1 = ((const float4*)g1)[0]; q1 = ((const float4*)g1)[1];
      p3 = ((const float4*)g3)[0]; q3 = ((const float4*)g3)[1];
      if (k0 + 64 < HD) {                                  // L2 warm-up
        __builtin_prefetch(g1 + (size_t)32 * ID, 0, 1);
        __builtin_prefetch(g3 + (size_t)32 * ID, 0, 1);
      }
      wait_async_le2();   // in-order: buffer `buf` asyncs complete
    } else {
      wait_async_le0();
    }
    __syncthreads();      // all waves' buf data visible (dscnt drained here)

    v16h a[2], b1f[2], b3f[2];
    a[0] = load_a_frag(sA[buf], wm);
    a[1] = load_a_frag(sA[buf], wm + 16);
    b1f[0] = load_b_frag(sB1[buf], wn);
    b1f[1] = load_b_frag(sB1[buf], wn + 16);
    b3f[0] = load_b_frag(sB3[buf], wn);
    b3f[1] = load_b_frag(sB3[buf], wn + 16);
#pragma unroll
    for (int mt = 0; mt < 2; ++mt)
#pragma unroll
      for (int nt = 0; nt < 2; ++nt) {
        accG[mt][nt] = wmma_f16(a[mt], b1f[nt], accG[mt][nt]);
        accU[mt][nt] = wmma_f16(a[mt], b3f[nt], accU[mt][nt]);
      }
    __syncthreads();      // protect buf from next iteration's writers
    buf = nxt;
  }

  // epilogue: silu(gate)*up -> f16 activation rows (slot-indexed, contiguous)
  const int em = (lane >> 4) * 8, en = lane & 15;
#pragma unroll
  for (int mt = 0; mt < 2; ++mt)
#pragma unroll
    for (int nt = 0; nt < 2; ++nt)
#pragma unroll
      for (int r = 0; r < 8; ++r) {
        int grow = m0 + wm + mt * 16 + em + r;
        if (grow < Me) {
          float g = accG[mt][nt][r];
          float u = accU[mt][nt][r];
          float s = g / (1.0f + __expf(-g));
          act[(size_t)(offe + grow) * ID + n0 + wn + nt * 16 + en] =
              (f16)(s * u);
        }
      }
}

// ---------------------------------------------------------------------------
// GEMM2: y = act @ W2_e ; rows contiguous by slot, output scattered by pair id.
// Same pipelined structure as GEMM1 with a single B matrix.
// ---------------------------------------------------------------------------
__global__ __launch_bounds__(256) void moe_ffn2(
    const f16* __restrict__ act, const float* __restrict__ w2,
    const int* __restrict__ tok_pid, const int* __restrict__ off,
    float* __restrict__ ybuf) {
  __shared__ f16 sA[2][128 * 32];
  __shared__ f16 sB[2][64 * 32];

  const int e = blockIdx.z;
  const int offe = off[e];
  const int Me = off[e + 1] - offe;
  const int m0 = blockIdx.y * 128;
  if (m0 >= Me) return;
  const int n0 = blockIdx.x * 64;

  const int tid = threadIdx.x, lane = tid & 31, wid = tid >> 5;
  const int wm = (wid & 3) * 32;
  const int wn = (wid >> 2) * 32;

  const float* W2 = w2 + (size_t)e * ID * HD;
  const int arow = tid >> 1, aseg = (tid & 1) * 16;
  // clamp padded rows to a valid slot: garbage only feeds discarded rows
  const int aslot = min(offe + m0 + arow, NT * TK - 1);
  const f16* aSrc = act + (size_t)aslot * ID + aseg;
  f16* aDst[2] = {&sA[0][arow * 32 + aseg], &sA[1][arow * 32 + aseg]};
  const int bk = tid >> 3, bn = (tid & 7) * 8;
  const float* B2 = W2 + (size_t)bk * HD + n0 + bn;

  v8f acc[2][2] = {};

  async_ld_b128(aDst[0], aSrc);
  async_ld_b128(aDst[0] + 8, aSrc + 8);
  float4 p = ((const float4*)B2)[0], q = ((const float4*)B2)[1];

  int buf = 0;
  for (int k0 = 0; k0 < ID; k0 += 32) {
    const int nxt = buf ^ 1;
    {
      f16 h[8] = {(f16)p.x, (f16)p.y, (f16)p.z, (f16)p.w,
                  (f16)q.x, (f16)q.y, (f16)q.z, (f16)q.w};
#pragma unroll
      for (int i = 0; i < 8; ++i) sB[buf][(bn + i) * 32 + bk] = h[i];
    }
    if (k0 + 32 < ID) {
      async_ld_b128(aDst[nxt], aSrc + (k0 + 32));
      async_ld_b128(aDst[nxt] + 8, aSrc + (k0 + 32) + 8);
      const float* g2 = B2 + (size_t)(k0 + 32) * HD;
      p = ((const float4*)g2)[0];
      q = ((const float4*)g2)[1];
      if (k0 + 64 < ID) __builtin_prefetch(g2 + (size_t)32 * HD, 0, 1);
      wait_async_le2();
    } else {
      wait_async_le0();
    }
    __syncthreads();

    v16h a[2], b[2];
    a[0] = load_a_frag(sA[buf], wm);
    a[1] = load_a_frag(sA[buf], wm + 16);
    b[0] = load_b_frag(sB[buf], wn);
    b[1] = load_b_frag(sB[buf], wn + 16);
#pragma unroll
    for (int mt = 0; mt < 2; ++mt)
#pragma unroll
      for (int nt = 0; nt < 2; ++nt)
        acc[mt][nt] = wmma_f16(a[mt], b[nt], acc[mt][nt]);
    __syncthreads();
    buf = nxt;
  }

  const int em = (lane >> 4) * 8, en = lane & 15;
#pragma unroll
  for (int mt = 0; mt < 2; ++mt)
#pragma unroll
    for (int nt = 0; nt < 2; ++nt)
#pragma unroll
      for (int r = 0; r < 8; ++r) {
        int grow = m0 + wm + mt * 16 + em + r;
        if (grow < Me) {
          int pid = tok_pid[offe + grow];  // scatter back to pair slot
          ybuf[(size_t)pid * HD + n0 + wn + nt * 16 + en] = acc[mt][nt][r];
        }
      }
}

// out[n] = ew[n,0]*y[pair(n,0)] + ew[n,1]*y[pair(n,1)]
__global__ void moe_combine(const float* __restrict__ ybuf,
                            const float* __restrict__ ew,
                            float* __restrict__ out) {
  int i = blockIdx.x * 256 + threadIdx.x;  // over N*H
  int n = i >> 10;                         // H == 1024
  int h = i & (HD - 1);
  out[i] = ew[2 * n] * ybuf[(size_t)(2 * n) * HD + h] +
           ew[2 * n + 1] * ybuf[(size_t)(2 * n + 1) * HD + h];
}

extern "C" void kernel_launch(void* const* d_in, const int* in_sizes, int n_in,
                              void* d_out, int out_size, void* d_ws,
                              size_t ws_size, hipStream_t stream) {
  const float* x = (const float*)d_in[0];
  const int* eidx = (const int*)d_in[1];
  const float* ew = (const float*)d_in[2];
  const float* w1 = (const float*)d_in[3];
  const float* w2 = (const float*)d_in[4];
  const float* w3 = (const float*)d_in[5];
  float* out = (float*)d_out;

  // workspace carve-up (~52 MB total)
  char* ws = (char*)d_ws;
  int* tok_pid = (int*)ws;                                   // 16 KB
  int* off = (int*)(ws + (16 << 10));                        // 64 B
  f16* xh = (f16*)(ws + (32 << 10));                         // 4 MB
  f16* act = (f16*)((char*)xh + (size_t)NT * HD * sizeof(f16));             // 32 MB
  float* ybuf = (float*)((char*)act + (size_t)NT * TK * ID * sizeof(f16));  // 16 MB

  moe_route<<<1, 256, 0, stream>>>(eidx, tok_pid, off);
  moe_cvt_x<<<(NT * HD / 4) / 256, 256, 0, stream>>>(x, xh);
  moe_ffn1<<<dim3(ID / 64, (NT * TK) / 128, NE), 256, 0, stream>>>(
      xh, w1, w3, tok_pid, off, act);
  moe_ffn2<<<dim3(HD / 64, (NT * TK) / 128, NE), 256, 0, stream>>>(
      act, w2, tok_pid, off, ybuf);
  moe_combine<<<(NT * HD) / 256, 256, 0, stream>>>(ybuf, ew, out);
}
